// NextTaskGAT_13469017441030
// MI455X (gfx1250) — compile-verified, matrix operand order checked by reading, not run on hardware
//
#include <hip/hip_runtime.h>
#include <hip/hip_bf16.h>
#include <math.h>

// ---------------- problem constants (match reference) ----------------
#define NN     20000
#define EE     320000
#define TOTE   (EE + NN)      // edges + self loops
#define DIN    128
#define HIDC   128
#define H1N    4
#define C1     (H1N * HIDC)   // 512
#define H2N    1
#define C2     (H2N * HIDC)   // 128
#define DOUT   32

typedef __attribute__((ext_vector_type(16))) _Float16 v16h;
typedef __attribute__((ext_vector_type(8)))  _Float16 v8h;
typedef __attribute__((ext_vector_type(8)))  float    v8f;

// ---------------- wave helpers (wave32, gfx1250) ----------------
__device__ __forceinline__ float wave_sum(float v) {
  #pragma unroll
  for (int m = 16; m > 0; m >>= 1) v += __shfl_xor(v, m, 32);
  return v;
}
__device__ __forceinline__ float wave_max(float v) {
  #pragma unroll
  for (int m = 16; m > 0; m >>= 1) v = fmaxf(v, __shfl_xor(v, m, 32));
  return v;
}
__device__ __forceinline__ void atomicMaxFloat(float* addr, float val) {
  // correct for all floats when addr initialized to -inf
  if (val >= 0.0f) atomicMax((int*)addr, __float_as_int(val));
  else             atomicMin((unsigned int*)addr, __float_as_uint(val));
}

// ---------------- weight repack: f32 [K,NCOLS] -> f16 WMMA-B-fragment-major ----------------
// packed index p = ((kchunk*(NCOLS/16) + ntile)*32 + lane)*16 + i
// lane holds column n = ntile*16 + (lane&15); element i maps to
// k_in_chunk = (i<8) ? (half*8+i) : (16+half*8+(i-8)), half = lane>>4.   (ISA 7.12.2)
template <int K, int NCOLS>
__global__ void repack_b_kernel(const float* __restrict__ W, _Float16* __restrict__ out) {
  int p = blockIdx.x * blockDim.x + threadIdx.x;
  if (p >= K * NCOLS) return;
  constexpr int NT = NCOLS / 16;
  int i    = p & 15;
  int lane = (p >> 4) & 31;
  int tile = (p >> 9) % NT;
  int chnk = (p >> 9) / NT;
  int half = lane >> 4;
  int n = tile * 16 + (lane & 15);
  int kin = (i < 8) ? (half * 8 + i) : (16 + half * 8 + (i - 8));
  int k = chnk * 32 + kin;
  out[p] = (_Float16)W[(long)k * NCOLS + n];
}

// ---------------- input LayerNorm (width 128) -> f16 ----------------
__global__ void ln128_f16_kernel(const float* __restrict__ x, const float* __restrict__ g,
                                 const float* __restrict__ b, _Float16* __restrict__ out, int nrows) {
  int row = blockIdx.x * (blockDim.x >> 5) + (threadIdx.x >> 5);
  if (row >= nrows) return;
  int lane = threadIdx.x & 31;
  const float4 xv = ((const float4*)(x + (long)row * 128))[lane];
  float mu = wave_sum(xv.x + xv.y + xv.z + xv.w) * (1.0f / 128.0f);
  float d0 = xv.x - mu, d1 = xv.y - mu, d2 = xv.z - mu, d3 = xv.w - mu;
  float var = wave_sum(d0 * d0 + d1 * d1 + d2 * d2 + d3 * d3) * (1.0f / 128.0f);
  float r = rsqrtf(var + 1e-5f);
  const float4 gv = ((const float4*)g)[lane];
  const float4 bv = ((const float4*)b)[lane];
  _Float16* o = out + (long)row * 128 + lane * 4;
  o[0] = (_Float16)(d0 * r * gv.x + bv.x);
  o[1] = (_Float16)(d1 * r * gv.y + bv.y);
  o[2] = (_Float16)(d2 * r * gv.z + bv.z);
  o[3] = (_Float16)(d3 * r * gv.w + bv.w);
}

// ---------------- LayerNorm + exact GELU ----------------
template <int WIDTH, typename OutT>
__global__ void ln_gelu_kernel(const float* __restrict__ in, const float* __restrict__ g,
                               const float* __restrict__ b, OutT* __restrict__ out, int nrows) {
  int row = blockIdx.x * (blockDim.x >> 5) + (threadIdx.x >> 5);
  if (row >= nrows) return;
  int lane = threadIdx.x & 31;
  constexpr int PER = WIDTH / 32;
  float v[PER];
  const float* r = in + (long)row * WIDTH;
  float s = 0.0f;
  #pragma unroll
  for (int i = 0; i < PER; i++) { v[i] = r[lane + i * 32]; s += v[i]; }
  float mu = wave_sum(s) * (1.0f / WIDTH);
  float q = 0.0f;
  #pragma unroll
  for (int i = 0; i < PER; i++) { v[i] -= mu; q += v[i] * v[i]; }
  float rstd = rsqrtf(wave_sum(q) * (1.0f / WIDTH) + 1e-5f);
  #pragma unroll
  for (int i = 0; i < PER; i++) {
    float y = v[i] * rstd * g[lane + i * 32] + b[lane + i * 32];
    float ge = 0.5f * y * (1.0f + erff(y * 0.70710678118654752f));
    out[(long)row * WIDTH + lane + i * 32] = (OutT)ge;
  }
}

// ---------------- WMMA GEMM: C[nrows,NCOLS] = A[nrows,K](f16) * Bpacked ----------------
// Each wave owns TPW adjacent 16x16 output tiles; A fragment loaded once per K-chunk
// (two 16B vector loads), each B fragment is two contiguous 16B vector loads.
template <int K, int NCOLS, int TPW>
__global__ void gemm_wmma_f16(const _Float16* __restrict__ A, const _Float16* __restrict__ Bp,
                              float* __restrict__ C, int nrows) {
  constexpr int NT = NCOLS / 16;
  int wave = threadIdx.x >> 5;
  int lane = threadIdx.x & 31;
  int m0 = blockIdx.x * 16;
  int t0 = (blockIdx.y * (blockDim.x >> 5) + wave) * TPW;   // first n-tile
  if (m0 >= nrows) return;
  v8f acc[TPW] = {};
  int half = lane >> 4;
  int m = lane & 15;
  const _Float16* Arow = A + (long)(m0 + m) * K + half * 8;
  #pragma unroll
  for (int kk = 0; kk < K; kk += 32) {
    const v8h alo = *(const v8h*)(Arow + kk);
    const v8h ahi = *(const v8h*)(Arow + kk + 16);
    v16h a;
    #pragma unroll
    for (int i = 0; i < 8; i++) { a[i] = alo[i]; a[8 + i] = ahi[i]; }
    const _Float16* bchunk = Bp + ((long)(kk >> 5) * NT) * 512 + lane * 16;
    #pragma unroll
    for (int t = 0; t < TPW; t++) {
      const _Float16* bp = bchunk + (long)(t0 + t) * 512;
      const v8h blo = *(const v8h*)(bp);
      const v8h bhi = *(const v8h*)(bp + 8);
      v16h b;
      #pragma unroll
      for (int i = 0; i < 8; i++) { b[i] = blo[i]; b[8 + i] = bhi[i]; }
      acc[t] = __builtin_amdgcn_wmma_f32_16x16x32_f16(false, a, false, b, (short)0, acc[t], false, false);
    }
  }
  int rbase = m0 + ((lane >> 4) ? 8 : 0);
  #pragma unroll
  for (int t = 0; t < TPW; t++) {
    int col = (t0 + t) * 16 + (lane & 15);
    #pragma unroll
    for (int j = 0; j < 8; j++) {
      int r = rbase + j;
      if (r < nrows) C[(long)r * NCOLS + col] = acc[t][j];
    }
  }
}

// ---------------- attention logits: al_s[n,h] = <xh[n,h,:], a_s[h,:]> ----------------
template <int H>
__global__ void att_logits_kernel(const float* __restrict__ xh, const float* __restrict__ a_s,
                                  const float* __restrict__ a_d, float* __restrict__ als,
                                  float* __restrict__ ald, int nrows) {
  int nh = blockIdx.x * (blockDim.x >> 5) + (threadIdx.x >> 5);
  if (nh >= nrows * H) return;
  int lane = threadIdx.x & 31;
  int h = nh % H;
  const float* row = xh + (long)nh * 128;   // [N, H*128] head-major
  float ss = 0.0f, sd = 0.0f;
  #pragma unroll
  for (int i = 0; i < 4; i++) {
    float xv = row[lane + i * 32];
    ss += xv * a_s[h * 128 + lane + i * 32];
    sd += xv * a_d[h * 128 + lane + i * 32];
  }
  ss = wave_sum(ss);
  sd = wave_sum(sd);
  if (lane == 0) { als[nh] = ss; ald[nh] = sd; }
}

// ---------------- init: out = bias (broadcast), m = -inf, den = 0 ----------------
__global__ void init_layer_kernel(float* __restrict__ out, const float* __restrict__ bias,
                                  int nrows, int width, float* __restrict__ m,
                                  float* __restrict__ den, int mcount) {
  int t = blockIdx.x * blockDim.x + threadIdx.x;
  long total = (long)nrows * width;
  if (t < total) out[t] = bias[t % width];
  if (t < mcount) { m[t] = -__builtin_inff(); den[t] = 0.0f; }
}

// ---------------- edge pass A: e = leaky_relu(al_s[src]+al_d[dst]); segment max ----------------
template <int H>
__global__ void edge_max_kernel(const long long* __restrict__ ei, const float* __restrict__ als,
                                const float* __restrict__ ald, float* __restrict__ ex,
                                float* __restrict__ m) {
  int t = blockIdx.x * blockDim.x + threadIdx.x;
  if (t >= TOTE * H) return;
  int e = t / H, h = t - e * H;
  int s, d;
  if (e < EE) { s = (int)ei[e]; d = (int)ei[EE + e]; } else { s = d = e - EE; }
  float v = als[s * H + h] + ald[d * H + h];
  v = (v > 0.0f) ? v : 0.2f * v;
  ex[t] = v;
  atomicMaxFloat(&m[d * H + h], v);
}

// ---------------- edge pass B: ex = exp(e - m[dst]); segment sum ----------------
template <int H>
__global__ void edge_exp_kernel(const long long* __restrict__ ei, float* __restrict__ ex,
                                const float* __restrict__ m, float* __restrict__ den) {
  int t = blockIdx.x * blockDim.x + threadIdx.x;
  if (t >= TOTE * H) return;
  int e = t / H, h = t - e * H;
  int d = (e < EE) ? (int)ei[EE + e] : (e - EE);
  float v = expf(ex[t] - m[d * H + h]);
  ex[t] = v;
  atomicAdd(&den[d * H + h], v);
}

// ---------------- edge pass C: out[dst] += alpha * xh[src]  (wave per edge) ----------------
template <int H>
__global__ void edge_scatter_kernel(const long long* __restrict__ ei, const float* __restrict__ xh,
                                    const float* __restrict__ ex, const float* __restrict__ den,
                                    float* __restrict__ out) {
  int e = blockIdx.x * (blockDim.x >> 5) + (threadIdx.x >> 5);
  if (e >= TOTE) return;
  int lane = threadIdx.x & 31;
  int s, d;
  if (e < EE) { s = (int)ei[e]; d = (int)ei[EE + e]; } else { s = d = e - EE; }
  constexpr int WIDTH = H * 128;
  #pragma unroll
  for (int it = 0; it < WIDTH / 128; it++) {
    int c = it * 128 + lane * 4;
    int h = c >> 7;
    float alpha = ex[e * H + h] / den[d * H + h];
    const float4 xv = *(const float4*)(xh + (long)s * WIDTH + c);
    float* o = out + (long)d * WIDTH + c;
    atomicAdd(o + 0, xv.x * alpha);
    atomicAdd(o + 1, xv.y * alpha);
    atomicAdd(o + 2, xv.z * alpha);
    atomicAdd(o + 3, xv.w * alpha);
  }
}

// ---------------- final dense 128->32 + bias + log_softmax (wave per node) ----------------
__global__ void final_kernel(const float* __restrict__ h, const float* __restrict__ Wo,
                             const float* __restrict__ bo, float* __restrict__ out, int nrows) {
  int row = blockIdx.x * (blockDim.x >> 5) + (threadIdx.x >> 5);
  if (row >= nrows) return;
  int lane = threadIdx.x & 31;
  float acc = bo[lane];
  const float* hr = h + (long)row * 128;
  #pragma unroll 8
  for (int k = 0; k < 128; k++) acc += hr[k] * Wo[k * 32 + lane];
  float mx = wave_max(acc);
  float ev = expf(acc - mx);
  float s = wave_sum(ev);
  out[(long)row * 32 + lane] = acc - mx - logf(s);
}

// =====================================================================
extern "C" void kernel_launch(void* const* d_in, const int* in_sizes, int n_in,
                              void* d_out, int out_size, void* d_ws, size_t ws_size,
                              hipStream_t stream) {
  const float*     x      = (const float*)d_in[0];
  const long long* ei     = (const long long*)d_in[1];
  const float*     g_in   = (const float*)d_in[2];
  const float*     b_in   = (const float*)d_in[3];
  const float*     W1     = (const float*)d_in[4];
  const float*     att1_s = (const float*)d_in[5];
  const float*     att1_d = (const float*)d_in[6];
  const float*     bias1  = (const float*)d_in[7];
  const float*     g1     = (const float*)d_in[8];
  const float*     b1     = (const float*)d_in[9];
  const float*     W2     = (const float*)d_in[10];
  const float*     att2_s = (const float*)d_in[11];
  const float*     att2_d = (const float*)d_in[12];
  const float*     bias2  = (const float*)d_in[13];
  const float*     g2     = (const float*)d_in[14];
  const float*     b2     = (const float*)d_in[15];
  const float*     Wo     = (const float*)d_in[16];
  const float*     bo     = (const float*)d_in[17];
  float* out = (float*)d_out;

  // ---- carve workspace ----
  char* wp = (char*)d_ws;
  auto carve = [&](size_t bytes) -> void* {
    void* p = (void*)wp;
    wp += (bytes + 255) & ~(size_t)255;
    return p;
  };
  _Float16* W1p  = (_Float16*)carve((size_t)DIN * C1 * 2);   // packed fragments
  _Float16* W2p  = (_Float16*)carve((size_t)C1 * C2 * 2);    // packed fragments
  _Float16* h0h  = (_Float16*)carve((size_t)NN * DIN * 2);
  float*    xh1  = (float*)   carve((size_t)NN * C1 * 4);
  float*    als1 = (float*)   carve((size_t)NN * H1N * 4);
  float*    ald1 = (float*)   carve((size_t)NN * H1N * 4);
  float*    m1   = (float*)   carve((size_t)NN * H1N * 4);
  float*    den1 = (float*)   carve((size_t)NN * H1N * 4);
  float*    ex1  = (float*)   carve((size_t)TOTE * H1N * 4);
  float*    out1 = (float*)   carve((size_t)NN * C1 * 4);
  _Float16* h1h  = (_Float16*)carve((size_t)NN * C1 * 2);
  float*    xh2  = (float*)   carve((size_t)NN * C2 * 4);
  float*    als2 = (float*)   carve((size_t)NN * H2N * 4);
  float*    ald2 = (float*)   carve((size_t)NN * H2N * 4);
  float*    m2   = (float*)   carve((size_t)NN * H2N * 4);
  float*    den2 = (float*)   carve((size_t)NN * H2N * 4);
  float*    ex2  = (float*)   carve((size_t)TOTE * H2N * 4);
  float*    out2 = (float*)   carve((size_t)NN * C2 * 4);
  float*    h2g  = (float*)   carve((size_t)NN * C2 * 4);

  const int BLK = 256;           // 8 waves / block
  const int WPB = BLK / 32;

  // weights -> packed f16 WMMA fragments
  repack_b_kernel<DIN, C1><<<(DIN * C1 + BLK - 1) / BLK, BLK, 0, stream>>>(W1, W1p);
  repack_b_kernel<C1, C2><<<(C1 * C2 + BLK - 1) / BLK, BLK, 0, stream>>>(W2, W2p);

  // LN(x) -> f16
  ln128_f16_kernel<<<(NN + WPB - 1) / WPB, BLK, 0, stream>>>(x, g_in, b_in, h0h, NN);

  // ---- layer 1 ----
  gemm_wmma_f16<DIN, C1, 2><<<dim3(NN / 16, C1 / (16 * WPB * 2)), BLK, 0, stream>>>(h0h, W1p, xh1, NN);
  att_logits_kernel<H1N><<<(NN * H1N + WPB - 1) / WPB, BLK, 0, stream>>>(xh1, att1_s, att1_d, als1, ald1, NN);
  {
    long total = (long)NN * C1;
    init_layer_kernel<<<(int)((total + BLK - 1) / BLK), BLK, 0, stream>>>(out1, bias1, NN, C1, m1, den1, NN * H1N);
  }
  edge_max_kernel<H1N><<<(TOTE * H1N + BLK - 1) / BLK, BLK, 0, stream>>>(ei, als1, ald1, ex1, m1);
  edge_exp_kernel<H1N><<<(TOTE * H1N + BLK - 1) / BLK, BLK, 0, stream>>>(ei, ex1, m1, den1);
  edge_scatter_kernel<H1N><<<(TOTE + WPB - 1) / WPB, BLK, 0, stream>>>(ei, xh1, ex1, den1, out1);
  ln_gelu_kernel<C1, _Float16><<<(NN + WPB - 1) / WPB, BLK, 0, stream>>>(out1, g1, b1, h1h, NN);

  // ---- layer 2 ----
  gemm_wmma_f16<C1, C2, 1><<<dim3(NN / 16, C2 / (16 * WPB)), BLK, 0, stream>>>(h1h, W2p, xh2, NN);
  att_logits_kernel<H2N><<<(NN * H2N + WPB - 1) / WPB, BLK, 0, stream>>>(xh2, att2_s, att2_d, als2, ald2, NN);
  {
    long total = (long)NN * C2;
    init_layer_kernel<<<(int)((total + BLK - 1) / BLK), BLK, 0, stream>>>(out2, bias2, NN, C2, m2, den2, NN * H2N);
  }
  edge_max_kernel<H2N><<<(TOTE * H2N + BLK - 1) / BLK, BLK, 0, stream>>>(ei, als2, ald2, ex2, m2);
  edge_exp_kernel<H2N><<<(TOTE * H2N + BLK - 1) / BLK, BLK, 0, stream>>>(ei, ex2, m2, den2);
  edge_scatter_kernel<H2N><<<(TOTE + WPB - 1) / WPB, BLK, 0, stream>>>(ei, xh2, ex2, den2, out2);
  ln_gelu_kernel<C2, float><<<(NN + WPB - 1) / WPB, BLK, 0, stream>>>(out2, g2, b2, h2g, NN);

  // ---- head ----
  final_kernel<<<(NN + WPB - 1) / WPB, BLK, 0, stream>>>(h2g, Wo, bo, out, NN);
}
